// DeepSeekMoE_60601988547207
// MI455X (gfx1250) — compile-verified, matrix operand order checked by reading, not run on hardware
//
#include <hip/hip_runtime.h>
#include <stdint.h>

// DeepSeek-style MoE (N=8192 tokens, C=1024, H=4096, 16 experts top-2 + 1 shared)
// MI455X (gfx1250, wave32) strategy:
//   - v_wmma_f32_16x16x32_bf16 for all four GEMMs, f32 accumulate
//   - A tiles (bf16) staged global->LDS via global_load_async_to_lds_b128 (ASYNCcnt)
//   - B tiles (fp32 weights) stream-converted to packed bf16 dwords, ds_store_b64
//   - fragments read as 2x ds_load_b128 per 16x16 tile (LDS pitch 144B, 16B aligned)
//   - grouped-GEMM over per-expert token lists; routing weight folded into GEMM2
//     epilogue with f32 atomicAdd combine on top of the shared-expert output.

#define N_TOK   8192
#define C_DIM   1024
#define H_DIM   4096
#define N_EXP   16

typedef __attribute__((ext_vector_type(16))) __bf16 v16bf;
typedef __attribute__((ext_vector_type(8)))  float  v8f;

union FragBF { v16bf v; uint4 q[2]; };
union AccU   { v8f   v; float f[8]; };

__device__ __forceinline__ uint32_t f2bf(float f) {
  uint32_t u = __builtin_bit_cast(uint32_t, f);
  u += 0x7FFFu + ((u >> 16) & 1u);   // round-to-nearest-even
  return u >> 16;
}
__device__ __forceinline__ uint32_t pk2bf(float lo, float hi) {
#if __has_builtin(__builtin_amdgcn_cvt_pk_bf16_f32)
  typedef __attribute__((ext_vector_type(2))) __bf16 v2bf;
  v2bf r = __builtin_amdgcn_cvt_pk_bf16_f32(lo, hi);
  return __builtin_bit_cast(uint32_t, r);
#else
  return f2bf(lo) | (f2bf(hi) << 16);
#endif
}

// LDS byte offset of a __shared__ object: generic pointer low 32 bits (ISA 10.2:
// flat->LDS address is addr[31:0]).
__device__ __forceinline__ uint32_t lds_off(const void* p) {
  return (uint32_t)(uintptr_t)p;
}

// CDNA5 async global->LDS copy (ASYNCcnt); offset applies to BOTH addresses.
#define ASYNC_B128(ldsaddr, gaddr, OFF)                                        \
  asm volatile("global_load_async_to_lds_b128 %0, %1, off offset:" #OFF        \
               :: "v"(ldsaddr), "v"(gaddr) : "memory")
#define WAIT_ASYNC() asm volatile("s_wait_asynccnt 0x0" ::: "memory")

// ---------------- elementwise cast: fp32 -> bf16 (8 elems / thread) ----------
__global__ __launch_bounds__(256) void cast_bf16_kernel(const float* __restrict__ in,
                                                        uint32_t* __restrict__ out,
                                                        int n8) {
  int i = blockIdx.x * 256 + threadIdx.x;
  if (i >= n8) return;
  const float4* p = (const float4*)in;
  float4 a = p[2 * i], b = p[2 * i + 1];
  uint4 o;
  o.x = pk2bf(a.x, a.y); o.y = pk2bf(a.z, a.w);
  o.z = pk2bf(b.x, b.y); o.w = pk2bf(b.z, b.w);
  ((uint4*)out)[i] = o;
}

// ---------------- router: 1 wave per token ----------------------------------
__global__ __launch_bounds__(256) void router_kernel(const float* __restrict__ x,
                                                     const float* __restrict__ rw,
                                                     const float* __restrict__ ebias,
                                                     int*   __restrict__ cnt,
                                                     int*   __restrict__ tokens,
                                                     float* __restrict__ tweights) {
  int t    = (blockIdx.x * blockDim.x + threadIdx.x) >> 5;
  int lane = threadIdx.x & 31;
  if (t >= N_TOK) return;
  const float* xr = x + (size_t)t * C_DIM;
  float acc[N_EXP];
#pragma unroll
  for (int e = 0; e < N_EXP; ++e) acc[e] = 0.f;
  for (int k = lane; k < C_DIM; k += 32) {
    float xv = xr[k];
    const float* r = rw + (size_t)k * N_EXP;
#pragma unroll
    for (int e = 0; e < N_EXP; ++e) acc[e] += xv * r[e];
  }
#pragma unroll
  for (int e = 0; e < N_EXP; ++e) {
#pragma unroll
    for (int off = 16; off; off >>= 1) acc[e] += __shfl_xor(acc[e], off, 32);
  }
  if (lane == 0) {
    float b1 = -1e30f, b2 = -1e30f; int i1 = 0, i2 = 0;
#pragma unroll
    for (int e = 0; e < N_EXP; ++e) {
      float v = acc[e] + ebias[e];
      if (v > b1)      { b2 = b1; i2 = i1; b1 = v; i1 = e; }
      else if (v > b2) { b2 = v;  i2 = e; }
    }
    float e2 = __expf(b2 - b1);
    float w1 = 1.f / (1.f + e2);
    float w2 = e2  / (1.f + e2);
    int p1 = atomicAdd(&cnt[i1], 1);
    tokens[i1 * N_TOK + p1] = t;  tweights[i1 * N_TOK + p1] = w1;
    int p2 = atomicAdd(&cnt[i2], 1);
    tokens[i2 * N_TOK + p2] = t;  tweights[i2 * N_TOK + p2] = w2;
  }
}

__global__ void init_counts_kernel(int* __restrict__ cnt) {
  if (threadIdx.x < N_EXP) cnt[threadIdx.x] = 0;
}

__global__ void scan_kernel(const int* __restrict__ cnt, int* __restrict__ offs) {
  if (threadIdx.x == 0) {
    int s = 0;
    for (int e = 0; e < N_EXP; ++e) { offs[e] = s; s += cnt[e]; }
  }
}

// ---------------- grouped GEMM, 128x128x64 tile, 8 waves --------------------
// MODE 0: shared GEMM1  : h = relu(xb @ w1 + b1)        -> bf16
// MODE 1: routed GEMM1  : gathered rows, same epilogue  -> bf16 (slot-indexed)
// MODE 2: shared GEMM2  : out = h @ w2 + b2             -> f32 plain store
// MODE 3: routed GEMM2  : out[token] += w * (h @ w2 + b2)  (atomicAdd f32)
template <int MODE, int K_DIM, int N_OUT>
__global__ __launch_bounds__(256) void moe_gemm_kernel(
    const uint16_t* __restrict__ A_base,   // bf16 activations [rows][K_DIM]
    const float*    __restrict__ W_base,   // fp32 weights [e][K_DIM][N_OUT]
    const float*    __restrict__ B_bias,   // fp32 bias [e][N_OUT]
    void*           __restrict__ out_base,
    const int*      __restrict__ cnt,
    const int*      __restrict__ offs,
    const int*      __restrict__ tokens,
    const float*    __restrict__ tweights,
    int M_total) {
  constexpr int BM = 128, BN = 128, BK = 64;
  constexpr int LD = BK + 8;   // 72 elems = 144 B/row: 16B-aligned, conflict-friendly
  __shared__ uint16_t As[BM][LD];
  __shared__ uint16_t Bs[BN][LD];
  __shared__ int   rowA[BM];
  __shared__ int   rowOutTok[BM];
  __shared__ float rowW[BM];

  const int e      = blockIdx.z;
  const int tile_n = blockIdx.x * BN;
  const int tile_m = blockIdx.y * BM;

  const int M = (MODE == 1 || MODE == 3) ? cnt[e] : M_total;
  if (tile_m >= M) return;   // load-imbalance early exit

  const float* W  = W_base + (size_t)e * K_DIM * N_OUT;
  const float* bb = B_bias + (size_t)e * N_OUT;

  const int tid = threadIdx.x;
  if (tid < BM) {
    int m  = tile_m + tid;
    int mc = (m < M) ? m : (M - 1);
    if (MODE == 0 || MODE == 2) {
      rowA[tid] = mc;
    } else if (MODE == 1) {
      rowA[tid] = tokens[e * N_TOK + mc];
    } else {
      rowA[tid]      = offs[e] + mc;
      rowOutTok[tid] = tokens[e * N_TOK + mc];
      rowW[tid]      = tweights[e * N_TOK + mc];
    }
  }
  __syncthreads();

  const int lane = tid & 31;
  const int wave = tid >> 5;
  const int wm   = wave >> 1;   // 0..3 (32 rows each)
  const int wn   = wave & 1;    // 0..1 (64 cols each)
  const int half = lane >> 4;
  const int l16  = lane & 15;

  // A staging: thread -> (row = tid/2, 32 bf16 = 64 bytes at (tid&1)*32), async
  const int arow = tid >> 1;
  const int acol = (tid & 1) * 32;
  const uint16_t* aptr   = A_base + (size_t)rowA[arow] * K_DIM + acol;
  const uint32_t  a_lds  = lds_off(&As[arow][acol]);

  // B staging: thread -> 4 k-rows (kb..kb+3) x 8 cols (n0..n0+7) fp32
  const int kb  = (tid >> 4) * 4;
  const int bn0 = (tid & 15) * 8;
  const float* wptr = W + (size_t)kb * N_OUT + tile_n + bn0;

  AccU acc[2][4];
#pragma unroll
  for (int mi = 0; mi < 2; ++mi)
#pragma unroll
    for (int ni = 0; ni < 4; ++ni)
      acc[mi][ni].v = (v8f){0.f, 0.f, 0.f, 0.f, 0.f, 0.f, 0.f, 0.f};

  for (int k0 = 0; k0 < K_DIM; k0 += BK) {
    // ---- A tile: 64 bytes/thread, 4x async b128 (LDS+global advance together)
    {
      uint64_t ga = (uint64_t)(uintptr_t)(aptr + k0);
      ASYNC_B128(a_lds, ga, 0);
      ASYNC_B128(a_lds, ga, 16);
      ASYNC_B128(a_lds, ga, 32);
      ASYNC_B128(a_lds, ga, 48);
    }
    // ---- B tile: fp32 -> packed bf16 dwords -> ds_store_b64 (transposed [n][k])
    {
      const float* gbase = wptr + (size_t)k0 * N_OUT;
      float rows[4][8];
#pragma unroll
      for (int j = 0; j < 4; ++j) {
        const float4* g = (const float4*)(gbase + (size_t)j * N_OUT);
        float4 lo = g[0], hi = g[1];
        rows[j][0] = lo.x; rows[j][1] = lo.y; rows[j][2] = lo.z; rows[j][3] = lo.w;
        rows[j][4] = hi.x; rows[j][5] = hi.y; rows[j][6] = hi.z; rows[j][7] = hi.w;
      }
#pragma unroll
      for (int n = 0; n < 8; ++n) {
        uint2 d;
        d.x = pk2bf(rows[0][n], rows[1][n]);   // k = kb+0, kb+1
        d.y = pk2bf(rows[2][n], rows[3][n]);   // k = kb+2, kb+3
        *(uint2*)&Bs[bn0 + n][kb] = d;         // 8B-aligned (kb%4==0, 144B pitch)
      }
      if (k0 + BK < K_DIM)
        __builtin_prefetch(wptr + (size_t)(k0 + BK) * N_OUT, 0, 1);
    }
    WAIT_ASYNC();
    __syncthreads();

    // ---- compute: 2 k-substeps of 32; fragments = 2x ds_load_b128 each
#pragma unroll
    for (int s = 0; s < 2; ++s) {
      FragBF a[2], b[4];
#pragma unroll
      for (int mi = 0; mi < 2; ++mi) {
        const uint4* p = (const uint4*)&As[wm * 32 + mi * 16 + l16][s * 32];
        a[mi].q[0] = p[half];
        a[mi].q[1] = p[2 + half];
      }
#pragma unroll
      for (int ni = 0; ni < 4; ++ni) {
        const uint4* p = (const uint4*)&Bs[wn * 64 + ni * 16 + l16][s * 32];
        b[ni].q[0] = p[half];
        b[ni].q[1] = p[2 + half];
      }
#pragma unroll
      for (int mi = 0; mi < 2; ++mi)
#pragma unroll
        for (int ni = 0; ni < 4; ++ni)
          acc[mi][ni].v = __builtin_amdgcn_wmma_f32_16x16x32_bf16(
              false, a[mi].v, false, b[ni].v, (short)0, acc[mi][ni].v, false, false);
    }
    __syncthreads();
  }

  // Epilogue: C layout = VGPR r -> row (r + 8*half), N = lane&15
#pragma unroll
  for (int mi = 0; mi < 2; ++mi) {
#pragma unroll
    for (int ni = 0; ni < 4; ++ni) {
      const int col = tile_n + wn * 64 + ni * 16 + l16;
      const float bv = bb[col];
#pragma unroll
      for (int r = 0; r < 8; ++r) {
        const int mLocal = wm * 32 + mi * 16 + half * 8 + r;
        const int m = tile_m + mLocal;
        if (m >= M) continue;
        float v = acc[mi][ni].f[r] + bv;
        if (MODE == 0) {
          float h = v > 0.f ? v : 0.f;
          ((uint16_t*)out_base)[(size_t)m * N_OUT + col] = (uint16_t)f2bf(h);
        } else if (MODE == 1) {
          float h = v > 0.f ? v : 0.f;
          ((uint16_t*)out_base)[(size_t)(offs[e] + (m)) * N_OUT + col] = (uint16_t)f2bf(h);
        } else if (MODE == 2) {
          ((float*)out_base)[(size_t)m * N_OUT + col] = v;
        } else {
          atomicAdd(&((float*)out_base)[(size_t)rowOutTok[mLocal] * N_OUT + col],
                    rowW[mLocal] * v);
        }
      }
    }
  }
}

// ---------------------------------------------------------------------------
extern "C" void kernel_launch(void* const* d_in, const int* in_sizes, int n_in,
                              void* d_out, int out_size, void* d_ws, size_t ws_size,
                              hipStream_t stream) {
  (void)in_sizes; (void)n_in; (void)out_size; (void)ws_size;
  const float* x           = (const float*)d_in[0];
  const float* router_w    = (const float*)d_in[1];
  const float* expert_bias = (const float*)d_in[2];
  const float* shared_w1   = (const float*)d_in[3];
  const float* shared_b1   = (const float*)d_in[4];
  const float* shared_w2   = (const float*)d_in[5];
  const float* shared_b2   = (const float*)d_in[6];
  const float* routed_w1   = (const float*)d_in[7];
  const float* routed_b1   = (const float*)d_in[8];
  const float* routed_w2   = (const float*)d_in[9];
  const float* routed_b2   = (const float*)d_in[10];
  float* out = (float*)d_out;

  // workspace carve-up (~219 MB)
  char* ws = (char*)d_ws;
  uint16_t* xb       = (uint16_t*)ws;  ws += (size_t)N_TOK * C_DIM * 2;       // 16 MB
  uint16_t* h_shared = (uint16_t*)ws;  ws += (size_t)N_TOK * H_DIM * 2;       // 67 MB
  uint16_t* h_routed = (uint16_t*)ws;  ws += (size_t)N_TOK * 2 * H_DIM * 2;   // 134 MB
  int*   cnt    = (int*)ws;    ws += N_EXP * sizeof(int);
  int*   offs   = (int*)ws;    ws += N_EXP * sizeof(int);
  int*   tokens = (int*)ws;    ws += (size_t)N_EXP * N_TOK * sizeof(int);
  float* twt    = (float*)ws;

  // 1) x -> bf16
  cast_bf16_kernel<<<(N_TOK * C_DIM / 8 + 255) / 256, 256, 0, stream>>>(
      x, (uint32_t*)xb, N_TOK * C_DIM / 8);

  // 2) router -> per-expert token lists
  init_counts_kernel<<<1, 32, 0, stream>>>(cnt);
  router_kernel<<<N_TOK / 8, 256, 0, stream>>>(x, router_w, expert_bias, cnt, tokens, twt);
  scan_kernel<<<1, 32, 0, stream>>>(cnt, offs);

  const dim3 blk(256);
  // 3) shared expert: GEMM1 (relu) then GEMM2 (writes out base value)
  moe_gemm_kernel<0, C_DIM, H_DIM><<<dim3(H_DIM / 128, N_TOK / 128, 1), blk, 0, stream>>>(
      xb, shared_w1, shared_b1, h_shared, nullptr, nullptr, nullptr, nullptr, N_TOK);
  moe_gemm_kernel<2, H_DIM, C_DIM><<<dim3(C_DIM / 128, N_TOK / 128, 1), blk, 0, stream>>>(
      h_shared, shared_w2, shared_b2, out, nullptr, nullptr, nullptr, nullptr, N_TOK);

  // 4) routed experts: grouped gather-GEMM1 then weighted scatter-GEMM2
  moe_gemm_kernel<1, C_DIM, H_DIM><<<dim3(H_DIM / 128, N_TOK / 128, N_EXP), blk, 0, stream>>>(
      xb, routed_w1, routed_b1, h_routed, cnt, offs, tokens, twt, 0);
  moe_gemm_kernel<3, H_DIM, C_DIM><<<dim3(C_DIM / 128, N_TOK / 128, N_EXP), blk, 0, stream>>>(
      h_routed, routed_w2, routed_b2, out, cnt, offs, tokens, twt, 0);
}
